// Switch_78735340471045
// MI455X (gfx1250) — matrix operand order repaired and checked by reading, not compiled, not run on hardware
//
#include <hip/hip_runtime.h>
#include <stdint.h>

typedef __attribute__((ext_vector_type(16))) __bf16 v16bf;
typedef __attribute__((ext_vector_type(8)))  __bf16 v8bf;
typedef __attribute__((ext_vector_type(2)))  __bf16 v2bf;
typedef __attribute__((ext_vector_type(8)))  float  v8f;
typedef __attribute__((ext_vector_type(4)))  float  v4f;

typedef __attribute__((address_space(3))) __bf16 lds_bf16;

static constexpr int NTOK = 8192;   // B*T
static constexpr int DDIM = 1024;
static constexpr int ENUM = 16;
static constexpr int HDIM = 4096;
static constexpr int CAP  = 640;    // int(N*1.25/E)
static constexpr float SIG = 0.1f;

// ---------------------------------------------------------------- init
__global__ __launch_bounds__(256) void k_init(float* __restrict__ y,
                                              int* __restrict__ s2t,
                                              int* __restrict__ cnt,
                                              float* __restrict__ stats) {
  int gid = blockIdx.x * 256 + threadIdx.x;
  v4f z4 = {0.f, 0.f, 0.f, 0.f};
  if (gid < NTOK * DDIM / 4) ((v4f*)y)[gid] = z4;       // zero combine output
  if (gid < ENUM * CAP) s2t[gid] = -1;
  if (gid < ENUM) cnt[gid] = 0;
  if (gid < 64) stats[gid] = 0.f;
}

// ---------------------------------------------------------------- router
// block = 256 threads = 16 tokens; lane-group of 16 = one expert per lane.
__global__ __launch_bounds__(256) void k_router(
    const float* __restrict__ x, const uint8_t* __restrict__ mask,
    const float* __restrict__ noise, const float* __restrict__ Wr,
    int* __restrict__ eid_out, float* __restrict__ gate_out,
    float* __restrict__ stats) {  // [0..15] p-sum, [16..31] f-cnt, [32] zz, [33] n
  const int t = threadIdx.x;
  const int token = blockIdx.x * 16 + (t >> 4);
  const int e = t & 15;

  const float* xr = x + (size_t)token * DDIM;
  float acc = 0.f;
  #pragma unroll 4
  for (int d = 0; d < DDIM; d += 4) {
    v4f xv = *(const v4f*)(xr + d);
    acc += xv[0] * Wr[(d + 0) * ENUM + e];
    acc += xv[1] * Wr[(d + 1) * ENUM + e];
    acc += xv[2] * Wr[(d + 2) * ENUM + e];
    acc += xv[3] * Wr[(d + 3) * ENUM + e];
  }
  const bool m = mask[token] != 0;
  const float nz = noise[(size_t)token * ENUM + e];
  const float mult = m ? (1.0f + (nz * 2.0f - 1.0f) * SIG) : 1.0f;
  const float ln = acc * mult;

  // ---- noisy softmax over the 16-lane group
  float mx = ln;
  #pragma unroll
  for (int s = 8; s; s >>= 1) mx = fmaxf(mx, __shfl_xor(mx, s, 16));
  float pe = expf(ln - mx);
  float sm = pe;
  #pragma unroll
  for (int s = 8; s; s >>= 1) sm += __shfl_xor(sm, s, 16);
  const float prob = pe / sm;
  float bv = ln; int bi = e;
  #pragma unroll
  for (int s = 8; s; s >>= 1) {
    float ov = __shfl_xor(bv, s, 16);
    int   oi = __shfl_xor(bi, s, 16);
    if (ov > bv || (ov == bv && oi < bi)) { bv = ov; bi = oi; }
  }
  const float gate = __shfl(prob, bi, 16);

  // ---- clean softmax (aux losses)
  float mc = acc;
  #pragma unroll
  for (int s = 8; s; s >>= 1) mc = fmaxf(mc, __shfl_xor(mc, s, 16));
  float pc = expf(acc - mc);
  float sc = pc;
  #pragma unroll
  for (int s = 8; s; s >>= 1) sc += __shfl_xor(sc, s, 16);
  const float probc = pc / sc;
  float bvc = acc; int bic = e;
  #pragma unroll
  for (int s = 8; s; s >>= 1) {
    float ov = __shfl_xor(bvc, s, 16);
    int   oi = __shfl_xor(bic, s, 16);
    if (ov > bvc || (ov == bvc && oi < bic)) { bvc = ov; bic = oi; }
  }
  const float z = mc + logf(sc);

  if (m) atomicAdd(&stats[e], probc);
  if (e == 0) {
    eid_out[token]  = bi;
    gate_out[token] = gate;
    if (m) {
      atomicAdd(&stats[16 + bic], 1.0f);
      atomicAdd(&stats[32], z * z);
      atomicAdd(&stats[33], 1.0f);
    }
  }
}

// ---------------------------------------------------------------- dispatch
__global__ __launch_bounds__(256) void k_compact(const uint8_t* __restrict__ mask,
                                                 const int* __restrict__ eid,
                                                 int* __restrict__ cnt,
                                                 int* __restrict__ lists) {
  int gid = blockIdx.x * 256 + threadIdx.x;
  if (gid >= NTOK) return;
  if (!mask[gid]) return;
  int e = eid[gid];
  int p = atomicAdd(&cnt[e], 1);
  lists[(size_t)e * NTOK + p] = gid;
}

// Exact rank within expert by (gate desc, token idx asc) -> deterministic slots.
__global__ __launch_bounds__(256) void k_rank(const int* __restrict__ cnt,
                                              const int* __restrict__ lists,
                                              const float* __restrict__ gate,
                                              int* __restrict__ s2t) {
  const int e = blockIdx.x;
  const int ne = cnt[e];
  const int* L = lists + (size_t)e * NTOK;
  for (int i = threadIdx.x; i < ne; i += 256) {
    const int ti = L[i];
    const float gi = gate[ti];
    int rank = 0;
    for (int j = 0; j < ne; ++j) {
      const int tj = L[j];
      const float gj = gate[tj];
      rank += (gj > gi) || (gj == gi && tj < ti);
    }
    if (rank < CAP) s2t[e * CAP + rank] = ti;
  }
}

__global__ __launch_bounds__(256) void k_gather(const float* __restrict__ x,
                                                const int* __restrict__ s2t,
                                                __bf16* __restrict__ bufA) {
  int gid = blockIdx.x * 256 + threadIdx.x;   // one thread per 8 elements
  int slot = gid >> 7;                         // DDIM/8 = 128 chunks per row
  int c = (gid & 127) << 3;
  int tok = s2t[slot];
  v8bf out;
  if (tok >= 0) {
    const float* xr = x + (size_t)tok * DDIM + c;
    v4f a = *(const v4f*)xr;
    v4f b = *(const v4f*)(xr + 4);
    out[0] = (__bf16)a[0]; out[1] = (__bf16)a[1];
    out[2] = (__bf16)a[2]; out[3] = (__bf16)a[3];
    out[4] = (__bf16)b[0]; out[5] = (__bf16)b[1];
    out[6] = (__bf16)b[2]; out[7] = (__bf16)b[3];
  } else {
    #pragma unroll
    for (int i = 0; i < 8; ++i) out[i] = (__bf16)0.f;
  }
  *(v8bf*)(bufA + (size_t)slot * DDIM + c) = out;
}

// ---------------------------------------------------------------- expert GEMM
// 128x128 block tile, 8 waves, wave tile 64x32 -> 8 wmma_f32_16x16x32_bf16/K-step.
// Double-buffered K pipeline: A (bf16) via global_load_async_to_lds_b128 (ASYNCcnt),
// B (fp32) via registers with packed bf16 ds_store_b32.
// EPI==1: C = relu(A@B) -> bf16 Hout.   EPI==2: scatter C*gate to token rows of Yout.
template <int EPI>
__global__ __launch_bounds__(256) void k_gemm(
    const __bf16* __restrict__ Aall,  // [E*CAP, K] bf16
    const float*  __restrict__ Ball,  // [E, K, NC] fp32 (converted on load)
    __bf16* __restrict__ Hout, float* __restrict__ Yout,
    const int* __restrict__ s2t, const float* __restrict__ gate,
    int K, int NC) {
  __shared__ __align__(16) __bf16 As[2][128][40];   // [row][k], 80B stride
  __shared__ __align__(16) __bf16 Bs[2][128][40];   // [n][k]  (B transposed)
  __shared__ int   s_tok[128];
  __shared__ float s_gate[128];

  const int t = threadIdx.x;
  const int e = blockIdx.z;
  const int rowBase = e * CAP + blockIdx.y * 128;
  const int n0 = blockIdx.x * 128;

  const __bf16* Ae = Aall + (size_t)rowBase * K;
  const float*  Be = Ball + (size_t)e * K * NC + n0;

  const int wid = t >> 5, lane = t & 31;
  const int wr = wid >> 2, wc = wid & 3;       // 2x4 wave grid
  const int half = lane >> 4, l16 = lane & 15;

  // A async-copy indices: 2 x 16B per thread per K-step
  const int ar = t >> 2;            // 0..63
  const int ac = (t & 3) << 3;      // 0,8,16,24
  // B load indices: 2 K-row pairs x 4 cols per thread per K-step
  const int kk2 = (t >> 5) << 1;    // 0,2,...,14
  const int c4  = t & 31;

  auto issueA = [&](int buf, int k0) {
    uint32_t l0 = (uint32_t)(uintptr_t)(lds_bf16*)&As[buf][ar][ac];
    uint32_t l1 = (uint32_t)(uintptr_t)(lds_bf16*)&As[buf][ar + 64][ac];
    unsigned long long g0 =
        (unsigned long long)(uintptr_t)(Ae + (size_t)ar * K + k0 + ac);
    unsigned long long g1 =
        (unsigned long long)(uintptr_t)(Ae + (size_t)(ar + 64) * K + k0 + ac);
    asm volatile("global_load_async_to_lds_b128 %0, %1, off"
                 :: "v"(l0), "v"(g0) : "memory");
    asm volatile("global_load_async_to_lds_b128 %0, %1, off"
                 :: "v"(l1), "v"(g1) : "memory");
  };
  auto loadB = [&](int k0, v4f* r) {
    #pragma unroll
    for (int p = 0; p < 2; ++p) {
      int k = k0 + p * 16 + kk2;
      r[p * 2 + 0] = *(const v4f*)(Be + (size_t)k * NC + c4 * 4);
      r[p * 2 + 1] = *(const v4f*)(Be + (size_t)(k + 1) * NC + c4 * 4);
    }
  };
  auto storeB = [&](int buf, const v4f* r) {
    #pragma unroll
    for (int p = 0; p < 2; ++p) {
      int k = p * 16 + kk2;
      #pragma unroll
      for (int i = 0; i < 4; ++i) {   // packed bf16 pair -> ds_store_b32
        v2bf pk;
        pk[0] = (__bf16)r[p * 2 + 0][i];
        pk[1] = (__bf16)r[p * 2 + 1][i];
        *(v2bf*)&Bs[buf][c4 * 4 + i][k] = pk;
      }
    }
  };

  v8f acc[4][2];
  const v8f z8 = {0.f, 0.f, 0.f, 0.f, 0.f, 0.f, 0.f, 0.f};
  #pragma unroll
  for (int mt = 0; mt < 4; ++mt)
    #pragma unroll
    for (int nt = 0; nt < 2; ++nt) acc[mt][nt] = z8;

  v4f breg[4];
  // ---- pipeline prologue: stage tile 0
  issueA(0, 0);
  loadB(0, breg);
  storeB(0, breg);
  asm volatile("s_wait_asynccnt 0x0" ::: "memory");
  __syncthreads();

  const int steps = K / 32;
  for (int kt = 0; kt < steps; ++kt) {
    const int cur = kt & 1, nxt = cur ^ 1;
    const bool more = (kt + 1) < steps;
    if (more) {                      // start next tile while computing this one
      issueA(nxt, (kt + 1) * 32);
      loadB((kt + 1) * 32, breg);
    }

    v16bf af[4], bfr[2];
    #pragma unroll
    for (int mt = 0; mt < 4; ++mt) {   // 16x32 A frag per ISA 16-bit layout
      const __bf16* ap = &As[cur][wr * 64 + mt * 16 + l16][0];
      v8bf lo = *(const v8bf*)(ap + half * 8);
      v8bf hi = *(const v8bf*)(ap + 16 + half * 8);
      #pragma unroll
      for (int i = 0; i < 8; ++i) { af[mt][i] = lo[i]; af[mt][i + 8] = hi[i]; }
    }
    #pragma unroll
    for (int nt = 0; nt < 2; ++nt) {   // 32x16 B frag per ISA 16-bit layout (same k order as A)
      const __bf16* bp = &Bs[cur][wc * 32 + nt * 16 + l16][0];
      v8bf lo = *(const v8bf*)(bp + half * 8);
      v8bf hi = *(const v8bf*)(bp + 16 + half * 8);
      #pragma unroll
      for (int i = 0; i < 8; ++i) { bfr[nt][i] = lo[i]; bfr[nt][i + 8] = hi[i]; }
    }
    #pragma unroll
    for (int mt = 0; mt < 4; ++mt)
      #pragma unroll
      for (int nt = 0; nt < 2; ++nt)
        acc[mt][nt] = __builtin_amdgcn_wmma_f32_16x16x32_bf16(
            false, af[mt], false, bfr[nt], (short)0, acc[mt][nt], false, false);

    if (more) storeB(nxt, breg);     // buffer nxt free since barrier of kt-1
    asm volatile("s_wait_asynccnt 0x0" ::: "memory");
    __syncthreads();
  }

  if constexpr (EPI == 2) {
    if (t < 128) {
      int tok = s2t[e * CAP + blockIdx.y * 128 + t];
      s_tok[t] = tok;
      s_gate[t] = (tok >= 0) ? gate[tok] : 0.f;
    }
    __syncthreads();
  }

  #pragma unroll
  for (int mt = 0; mt < 4; ++mt)
    #pragma unroll
    for (int nt = 0; nt < 2; ++nt) {
      const int col = n0 + wc * 32 + nt * 16 + l16;
      #pragma unroll
      for (int r = 0; r < 8; ++r) {
        const int lrow = wr * 64 + mt * 16 + half * 8 + r;  // ISA C layout
        const float v = acc[mt][nt][r];
        if constexpr (EPI == 1) {
          Hout[(size_t)(rowBase + lrow) * NC + col] = (__bf16)fmaxf(v, 0.f);
        } else {
          const int tok = s_tok[lrow];
          if (tok >= 0) Yout[(size_t)tok * DDIM + col] = v * s_gate[lrow];
        }
      }
    }
}

// ---------------------------------------------------------------- losses
__global__ void k_final(const float* __restrict__ stats, float* __restrict__ out) {
  if (threadIdx.x == 0) {
    const float n = stats[33];
    float lb = 0.f;
    for (int e = 0; e < ENUM; ++e) lb += (stats[e] / n) * (stats[16 + e] / n);
    out[(size_t)NTOK * DDIM]     = (float)ENUM * lb;
    out[(size_t)NTOK * DDIM + 1] = stats[32] / n;
  }
}

// ---------------------------------------------------------------- launch
extern "C" void kernel_launch(void* const* d_in, const int* in_sizes, int n_in,
                              void* d_out, int out_size, void* d_ws, size_t ws_size,
                              hipStream_t stream) {
  (void)in_sizes; (void)n_in; (void)out_size; (void)ws_size;
  const float*   x     = (const float*)d_in[0];
  const uint8_t* mask  = (const uint8_t*)d_in[1];
  const float*   noise = (const float*)d_in[2];
  const float*   Wr    = (const float*)d_in[3];
  const float*   W1    = (const float*)d_in[4];
  const float*   W2    = (const float*)d_in[5];
  float* out = (float*)d_out;

  uint8_t* w = (uint8_t*)d_ws;
  auto carve = [&](size_t bytes) {
    uint8_t* p = w;
    w += (bytes + 255) & ~(size_t)255;
    return p;
  };
  int*    s2t   = (int*)carve((size_t)ENUM * CAP * 4);
  int*    cnt   = (int*)carve(ENUM * 4);
  int*    lists = (int*)carve((size_t)ENUM * NTOK * 4);
  int*    eid   = (int*)carve(NTOK * 4);
  float*  gate  = (float*)carve(NTOK * 4);
  float*  stats = (float*)carve(64 * 4);
  __bf16* bufA  = (__bf16*)carve((size_t)ENUM * CAP * DDIM * 2);
  __bf16* hbuf  = (__bf16*)carve((size_t)ENUM * CAP * HDIM * 2);

  k_init<<<NTOK * DDIM / 4 / 256, 256, 0, stream>>>(out, s2t, cnt, stats);
  k_router<<<NTOK / 16, 256, 0, stream>>>(x, mask, noise, Wr, eid, gate, stats);
  k_compact<<<NTOK / 256, 256, 0, stream>>>(mask, eid, cnt, lists);
  k_rank<<<ENUM, 256, 0, stream>>>(cnt, lists, gate, s2t);
  k_gather<<<(ENUM * CAP * DDIM / 8) / 256, 256, 0, stream>>>(x, s2t, bufA);
  k_gemm<1><<<dim3(HDIM / 128, CAP / 128, ENUM), 256, 0, stream>>>(
      bufA, W1, hbuf, nullptr, nullptr, nullptr, DDIM, HDIM);
  k_gemm<2><<<dim3(DDIM / 128, CAP / 128, ENUM), 256, 0, stream>>>(
      hbuf, W2, nullptr, out, s2t, gate, HDIM, DDIM);
  k_final<<<1, 32, 0, stream>>>(stats, out);
}
